// DSS_64364379898216
// MI455X (gfx1250) — compile-verified, matrix operand order checked by reading, not run on hardware
//
#include <hip/hip_runtime.h>
#include <hip/hip_bf16.h>

typedef __attribute__((ext_vector_type(16))) _Float16 v16h;
typedef __attribute__((ext_vector_type(8)))  _Float16 v8h;
typedef __attribute__((ext_vector_type(8)))  float    v8f;

#define D64 64
#define MIND 4
#define NHEAD 2

// half-weight offsets inside workspace (element units)
#define OFF_WIK  0
#define OFF_WIV  4096
#define OFF_WIO  8192
#define OFF_WIF1 12288
#define OFF_WIF2 16384
#define OFF_WMQ  20480
#define OFF_WMK  24576
#define OFF_WMV  28672
#define OFF_WMO  32768
#define OFF_WMF1 36864
#define OFF_WMF2 40960
#define N_WHALF  45056

__device__ __forceinline__ v8f zero8() {
  v8f z = {0.f,0.f,0.f,0.f,0.f,0.f,0.f,0.f};
  return z;
}

__device__ __forceinline__ v8f wmma16(v16h a, v16h b, v8f c) {
  return __builtin_amdgcn_wmma_f32_16x16x32_f16(false, a, false, b, (short)0, c, false, false);
}

// A-fragment (16x32, row-major, row=lane&15) or B-fragment of W^T (load W rows row-major).
__device__ __forceinline__ v16h load_frag_h(const _Float16* p, int ld) {
  const int lane = threadIdx.x & 31;
  const _Float16* r = p + (lane & 15) * ld + ((lane & 16) ? 8 : 0);
  v8h lo = *(const v8h*)r;
  v8h hi = *(const v8h*)(r + 16);
  v16h f;
#pragma unroll
  for (int i = 0; i < 8; ++i) { f[i] = lo[i]; f[i + 8] = hi[i]; }
  return f;
}

// D/C 16x16 f32 tile store: VGPR r -> row r (lanes 0-15) / row 8+r (lanes 16-31), col = lane&15
__device__ __forceinline__ void store_d_f32(float* p, int ld, v8f d) {
  const int lane = threadIdx.x & 31;
  const int col = lane & 15, rb = (lane & 16) ? 8 : 0;
#pragma unroll
  for (int r = 0; r < 8; ++r) p[(rb + r) * ld + col] = d[r];
}

__device__ __forceinline__ void store_d_h(_Float16* p, int ld, v8f d) {
  const int lane = threadIdx.x & 31;
  const int col = lane & 15, rb = (lane & 16) ? 8 : 0;
#pragma unroll
  for (int r = 0; r < 8; ++r) p[(rb + r) * ld + col] = (_Float16)d[r];
}

__device__ __forceinline__ float dot8(const float4 a0, const float4 a1, const v8h v) {
  return a0.x * (float)v[0] + a0.y * (float)v[1] + a0.z * (float)v[2] + a0.w * (float)v[3] +
         a1.x * (float)v[4] + a1.y * (float)v[5] + a1.z * (float)v[6] + a1.w * (float)v[7];
}

// ---------------- prep: f32 weights -> f16 workspace, Qi = bimp @ wi_q^T ----------------
__global__ void isab_prep(const float* wiK, const float* wiV, const float* wiO,
                          const float* wiF1, const float* wiF2,
                          const float* wmQ, const float* wmK, const float* wmV,
                          const float* wmO, const float* wmF1, const float* wmF2,
                          const float* wiQ, const float* bimp,
                          _Float16* wH, float* Qi) {
  int idx = blockIdx.x * 256 + threadIdx.x;
  if (idx < N_WHALF) {
    const float* srcs[11] = {wiK, wiV, wiO, wiF1, wiF2, wmQ, wmK, wmV, wmO, wmF1, wmF2};
    int m = idx >> 12, e = idx & 4095;
    wH[m * 4096 + e] = (_Float16)srcs[m][e];
  } else if (idx < N_WHALF + MIND * D64) {
    int t = idx - N_WHALF;
    int q = t >> 6, c = t & 63;
    float s = 0.f;
    for (int k = 0; k < D64; ++k) s += bimp[q * D64 + k] * wiQ[c * D64 + k];
    Qi[t] = s;
  }
}

// ---------------- main: 1 block (128 thr / 4 waves) = 16 bundles ----------------
__global__ void __launch_bounds__(128)
isab_kernel(const float* __restrict__ items, const float* __restrict__ bundles,
            const float* __restrict__ bimp, const float* __restrict__ Qi,
            const _Float16* __restrict__ wH, const int* __restrict__ bitems,
            const float* wi_ln1g, const float* wi_ln1b,
            const float* wi_ln2g, const float* wi_ln2b,
            const float* wi_f1b, const float* wi_f2b,
            const float* wm_ln1g, const float* wm_ln1b,
            const float* wm_ln2g, const float* wm_ln2b,
            const float* wm_f1b, const float* wm_f2b,
            float* __restrict__ out, int NB, int Kk) {
  __shared__ _Float16 sKh[4][32 * D64];   // per-wave Kh row-major (phase2: K2 [64][64])
  __shared__ _Float16 sVt[4][D64 * 32];   // per-wave V TRANSPOSED [col][key] (phase2: V2 row-major)
  __shared__ _Float16 sAtt[4][16 * D64];  // per-wave att-out / FF T (phase2: A2)
  __shared__ float    sH1[4][16 * D64];   // per-wave H f32 (phase2: Q2 / O2 / H2)
  __shared__ _Float16 sHh[4][16 * D64];   // per-wave H f16 (phase2: H2h / T2)
  __shared__ _Float16 sHind[64 * D64];    // block Hind [16 bundles x 4, 64]
  __shared__ float    sAttW[4][2 * 4 * 32];
  __shared__ float    sSc[16][2][4];

  const int wid  = threadIdx.x >> 5;
  const int lane = threadIdx.x & 31;
  const int blockBase = blockIdx.x * 16;
  const int col16 = lane & 15;
  const int rb = (lane & 16) ? 8 : 0;
  const float sc = 0.17677669529663687f; // 1/sqrt(32)

  _Float16* Khw = sKh[wid];
  _Float16* Vtw = sVt[wid];
  _Float16* Attw = sAtt[wid];
  float* H1w = sH1[wid];
  _Float16* Hhw = sHh[wid];
  float* attnW = sAttW[wid];

  // =================== Phase 1: MAB(Xi, Y) per bundle ===================
  for (int j = 0; j < 4; ++j) {
    int b = blockBase + wid * 4 + j;
    if (b >= NB) b = NB - 1;

    // gather Y into WMMA A fragments: unconditional vector loads + mask multiply
    v16h ay[2][2];
#pragma unroll
    for (int rt = 0; rt < 2; ++rt) {
      int key = rt * 16 + col16;
      int item = (key < Kk) ? bitems[b * Kk + key] : -1;
      bool valid = item >= 0;
      const float4* yr = (const float4*)(items + (long long)(valid ? item : 0) * D64);
      const int q0 = (lane & 16) ? 2 : 0;  // c0/4
      float4 p0 = yr[q0 + 0],  p1 = yr[q0 + 1];   // cols c0   .. c0+7
      float4 p2 = yr[q0 + 4],  p3 = yr[q0 + 5];   // cols c0+16.. c0+23
      float4 p4 = yr[q0 + 8],  p5 = yr[q0 + 9];   // cols c0+32.. c0+39
      float4 p6 = yr[q0 + 12], p7 = yr[q0 + 13];  // cols c0+48.. c0+55
      float mm = valid ? 1.f : 0.f;
      v16h f0, f1;
      f0[0]  = (_Float16)(p0.x * mm); f0[1]  = (_Float16)(p0.y * mm);
      f0[2]  = (_Float16)(p0.z * mm); f0[3]  = (_Float16)(p0.w * mm);
      f0[4]  = (_Float16)(p1.x * mm); f0[5]  = (_Float16)(p1.y * mm);
      f0[6]  = (_Float16)(p1.z * mm); f0[7]  = (_Float16)(p1.w * mm);
      f0[8]  = (_Float16)(p2.x * mm); f0[9]  = (_Float16)(p2.y * mm);
      f0[10] = (_Float16)(p2.z * mm); f0[11] = (_Float16)(p2.w * mm);
      f0[12] = (_Float16)(p3.x * mm); f0[13] = (_Float16)(p3.y * mm);
      f0[14] = (_Float16)(p3.z * mm); f0[15] = (_Float16)(p3.w * mm);
      f1[0]  = (_Float16)(p4.x * mm); f1[1]  = (_Float16)(p4.y * mm);
      f1[2]  = (_Float16)(p4.z * mm); f1[3]  = (_Float16)(p4.w * mm);
      f1[4]  = (_Float16)(p5.x * mm); f1[5]  = (_Float16)(p5.y * mm);
      f1[6]  = (_Float16)(p5.z * mm); f1[7]  = (_Float16)(p5.w * mm);
      f1[8]  = (_Float16)(p6.x * mm); f1[9]  = (_Float16)(p6.y * mm);
      f1[10] = (_Float16)(p6.z * mm); f1[11] = (_Float16)(p6.w * mm);
      f1[12] = (_Float16)(p7.x * mm); f1[13] = (_Float16)(p7.y * mm);
      f1[14] = (_Float16)(p7.z * mm); f1[15] = (_Float16)(p7.w * mm);
      ay[rt][0] = f0;
      ay[rt][1] = f1;
    }

    // Kh = Y @ wiK^T (row-major), V = Y @ wiV^T (transposed, packed b128 stores)
#pragma unroll
    for (int rt = 0; rt < 2; ++rt)
#pragma unroll
      for (int nt = 0; nt < 4; ++nt) {
        v8f aK = zero8(), aV = zero8();
#pragma unroll
        for (int kc = 0; kc < 2; ++kc) {
          aK = wmma16(ay[rt][kc], load_frag_h(wH + OFF_WIK + nt * 16 * D64 + kc * 32, D64), aK);
          aV = wmma16(ay[rt][kc], load_frag_h(wH + OFF_WIV + nt * 16 * D64 + kc * 32, D64), aV);
        }
        store_d_h(Khw + (rt * 16) * D64 + nt * 16, D64, aK);
        v8h pk;
#pragma unroll
        for (int r = 0; r < 8; ++r) pk[r] = (_Float16)aV[r];
        *(v8h*)(Vtw + (nt * 16 + col16) * 32 + rt * 16 + rb) = pk;
      }

    // scores + softmax (lane = key), vectorized LDS reads
    {
      int key = lane;
      int item = (key < Kk) ? bitems[b * Kk + key] : -1;
      bool valid = item >= 0;
      for (int h = 0; h < NHEAD; ++h)
        for (int q = 0; q < MIND; ++q) {
          const float4* qp = (const float4*)(Qi + q * D64 + h * 32);
          const v8h* kp = (const v8h*)(Khw + key * D64 + h * 32);
          float s = 0.f;
#pragma unroll
          for (int cb = 0; cb < 4; ++cb)
            s += dot8(qp[cb * 2], qp[cb * 2 + 1], kp[cb]);
          s *= sc;
          float m = valid ? s : -3.4e38f;
          for (int off = 16; off; off >>= 1) m = fmaxf(m, __shfl_xor(m, off, 32));
          float e = valid ? __expf(s - m) : 0.f;
          float se = e;
          for (int off = 16; off; off >>= 1) se += __shfl_xor(se, off, 32);
          attnW[h * 128 + q * 32 + key] = e / se;
        }
    }

    // att-out rows via transposed V: contiguous v8h reads along key
    for (int q = 0; q < MIND; ++q)
#pragma unroll
      for (int cc = 0; cc < 2; ++cc) {
        int col = lane + cc * 32;
        int h = col >> 5;
        const v8h* vp = (const v8h*)(Vtw + col * 32);
        const float4* ap = (const float4*)(attnW + h * 128 + q * 32);
        float acc = 0.f;
#pragma unroll
        for (int kb = 0; kb < 4; ++kb)  // keys >= Kk have attn == 0 and V row == 0
          acc += dot8(ap[kb * 2], ap[kb * 2 + 1], vp[kb]);
        Attw[(j * 4 + q) * D64 + col] = (_Float16)acc;
      }
  }

  // --- O = AttOut @ wiO^T + Xi -> LN1 -> H ---
  {
    v16h a0 = load_frag_h(Attw, D64);
    v16h a1 = load_frag_h(Attw + 32, D64);
#pragma unroll
    for (int nt = 0; nt < 4; ++nt) {
      v8f acc = zero8();
      acc = wmma16(a0, load_frag_h(wH + OFF_WIO + nt * 16 * D64, D64), acc);
      acc = wmma16(a1, load_frag_h(wH + OFF_WIO + nt * 16 * D64 + 32, D64), acc);
#pragma unroll
      for (int r = 0; r < 8; ++r) {
        int row = rb + r;
        float v = acc[r] + bimp[(row & 3) * D64 + nt * 16 + col16];
        H1w[row * D64 + nt * 16 + col16] = v;
      }
    }
  }
  if (lane < 16) {
    int row = lane;
    const float4* hp = (const float4*)(H1w + row * D64);
    float mu = 0.f;
#pragma unroll
    for (int cb = 0; cb < 16; ++cb) { float4 v = hp[cb]; mu += v.x + v.y + v.z + v.w; }
    mu *= (1.f / D64);
    float var = 0.f;
#pragma unroll
    for (int cb = 0; cb < 16; ++cb) {
      float4 v = hp[cb];
      var += (v.x-mu)*(v.x-mu) + (v.y-mu)*(v.y-mu) + (v.z-mu)*(v.z-mu) + (v.w-mu)*(v.w-mu);
    }
    float rs = rsqrtf(var * (1.f / D64) + 1e-5f);
#pragma unroll
    for (int cb = 0; cb < 8; ++cb) {
      v8h pk;
#pragma unroll
      for (int i = 0; i < 8; ++i) {
        int c = cb * 8 + i;
        float v = (H1w[row * D64 + c] - mu) * rs * wi_ln1g[c] + wi_ln1b[c];
        H1w[row * D64 + c] = v;
        pk[i] = (_Float16)v;
      }
      *(v8h*)(Hhw + row * D64 + cb * 8) = pk;
    }
  }

  // --- FF: relu(H@f1^T+b1)@f2^T+b2 + H -> LN2 -> Hind ---
  {
    v16h a0 = load_frag_h(Hhw, D64), a1 = load_frag_h(Hhw + 32, D64);
#pragma unroll
    for (int nt = 0; nt < 4; ++nt) {
      v8f acc = zero8();
      acc = wmma16(a0, load_frag_h(wH + OFF_WIF1 + nt * 16 * D64, D64), acc);
      acc = wmma16(a1, load_frag_h(wH + OFF_WIF1 + nt * 16 * D64 + 32, D64), acc);
      float bb = wi_f1b[nt * 16 + col16];
#pragma unroll
      for (int r = 0; r < 8; ++r)
        Attw[(rb + r) * D64 + nt * 16 + col16] = (_Float16)fmaxf(acc[r] + bb, 0.f);
    }
    v16h t0 = load_frag_h(Attw, D64), t1 = load_frag_h(Attw + 32, D64);
#pragma unroll
    for (int nt = 0; nt < 4; ++nt) {
      v8f acc = zero8();
      acc = wmma16(t0, load_frag_h(wH + OFF_WIF2 + nt * 16 * D64, D64), acc);
      acc = wmma16(t1, load_frag_h(wH + OFF_WIF2 + nt * 16 * D64 + 32, D64), acc);
      float bb = wi_f2b[nt * 16 + col16];
#pragma unroll
      for (int r = 0; r < 8; ++r) {
        int row = rb + r;
        H1w[row * D64 + nt * 16 + col16] = acc[r] + bb + H1w[row * D64 + nt * 16 + col16];
      }
    }
  }
  if (lane < 16) {
    int row = lane;
    const float4* hp = (const float4*)(H1w + row * D64);
    float mu = 0.f;
#pragma unroll
    for (int cb = 0; cb < 16; ++cb) { float4 v = hp[cb]; mu += v.x + v.y + v.z + v.w; }
    mu *= (1.f / D64);
    float var = 0.f;
#pragma unroll
    for (int cb = 0; cb < 16; ++cb) {
      float4 v = hp[cb];
      var += (v.x-mu)*(v.x-mu) + (v.y-mu)*(v.y-mu) + (v.z-mu)*(v.z-mu) + (v.w-mu)*(v.w-mu);
    }
    float rs = rsqrtf(var * (1.f / D64) + 1e-5f);
#pragma unroll
    for (int cb = 0; cb < 8; ++cb) {
      v8h pk;
#pragma unroll
      for (int i = 0; i < 8; ++i) {
        int c = cb * 8 + i;
        float v = (H1w[row * D64 + c] - mu) * rs * wi_ln2g[c] + wi_ln2b[c];
        pk[i] = (_Float16)v;
      }
      *(v8h*)(sHind + (wid * 16 + row) * D64 + cb * 8) = pk;
    }
  }
  __syncthreads();

  // =================== Phase 2: MAB(X, Hind), 16 bundles ===================
  _Float16* K2 = &sKh[0][0]; // [64][64] row-major
  _Float16* V2 = &sVt[0][0]; // [64][64] row-major (phase-2 reuse)
  float* Q2 = sH1[0];
  {
    v16h a0 = load_frag_h(sHind + wid * 16 * D64, D64);
    v16h a1 = load_frag_h(sHind + wid * 16 * D64 + 32, D64);
#pragma unroll
    for (int nt = 0; nt < 4; ++nt) {
      v8f aK = zero8(), aV = zero8();
      aK = wmma16(a0, load_frag_h(wH + OFF_WMK + nt * 16 * D64, D64), aK);
      aK = wmma16(a1, load_frag_h(wH + OFF_WMK + nt * 16 * D64 + 32, D64), aK);
      aV = wmma16(a0, load_frag_h(wH + OFF_WMV + nt * 16 * D64, D64), aV);
      aV = wmma16(a1, load_frag_h(wH + OFF_WMV + nt * 16 * D64 + 32, D64), aV);
      store_d_h(K2 + (wid * 16) * D64 + nt * 16, D64, aK);
      store_d_h(V2 + (wid * 16) * D64 + nt * 16, D64, aV);
    }
    // Q2 = X @ wmQ^T : each wave one n-tile
    int row = blockBase + col16; if (row >= NB) row = NB - 1;
    const float4* xr = (const float4*)(bundles + (long long)row * D64);
    const int q0 = (lane & 16) ? 2 : 0;
    float4 p0 = xr[q0 + 0],  p1 = xr[q0 + 1];
    float4 p2 = xr[q0 + 4],  p3 = xr[q0 + 5];
    float4 p4 = xr[q0 + 8],  p5 = xr[q0 + 9];
    float4 p6 = xr[q0 + 12], p7 = xr[q0 + 13];
    v16h x0, x1;
    x0[0]=(_Float16)p0.x; x0[1]=(_Float16)p0.y; x0[2]=(_Float16)p0.z; x0[3]=(_Float16)p0.w;
    x0[4]=(_Float16)p1.x; x0[5]=(_Float16)p1.y; x0[6]=(_Float16)p1.z; x0[7]=(_Float16)p1.w;
    x0[8]=(_Float16)p2.x; x0[9]=(_Float16)p2.y; x0[10]=(_Float16)p2.z; x0[11]=(_Float16)p2.w;
    x0[12]=(_Float16)p3.x; x0[13]=(_Float16)p3.y; x0[14]=(_Float16)p3.z; x0[15]=(_Float16)p3.w;
    x1[0]=(_Float16)p4.x; x1[1]=(_Float16)p4.y; x1[2]=(_Float16)p4.z; x1[3]=(_Float16)p4.w;
    x1[4]=(_Float16)p5.x; x1[5]=(_Float16)p5.y; x1[6]=(_Float16)p5.z; x1[7]=(_Float16)p5.w;
    x1[8]=(_Float16)p6.x; x1[9]=(_Float16)p6.y; x1[10]=(_Float16)p6.z; x1[11]=(_Float16)p6.w;
    x1[12]=(_Float16)p7.x; x1[13]=(_Float16)p7.y; x1[14]=(_Float16)p7.z; x1[15]=(_Float16)p7.w;
    int nt = wid;
    v8f acc = zero8();
    acc = wmma16(x0, load_frag_h(wH + OFF_WMQ + nt * 16 * D64, D64), acc);
    acc = wmma16(x1, load_frag_h(wH + OFF_WMQ + nt * 16 * D64 + 32, D64), acc);
    store_d_f32(Q2 + nt * 16, D64, acc);
  }
  __syncthreads();

  // scores (4 keys per bundle, per head)
  {
    int t = threadIdx.x;
    int b = t >> 3, h = (t >> 2) & 1, jj = t & 3;
    int krow = b * 4 + jj;
    const float4* qp = (const float4*)(Q2 + b * D64 + h * 32);
    const v8h* kp = (const v8h*)(K2 + krow * D64 + h * 32);
    float s = 0.f;
#pragma unroll
    for (int cb = 0; cb < 4; ++cb)
      s += dot8(qp[cb * 2], qp[cb * 2 + 1], kp[cb]);
    sSc[b][h][jj] = s * sc;
  }
  __syncthreads();
  if (threadIdx.x < 32) {
    int b = threadIdx.x >> 1, h = threadIdx.x & 1;
    float s0 = sSc[b][h][0], s1 = sSc[b][h][1], s2 = sSc[b][h][2], s3 = sSc[b][h][3];
    float m = fmaxf(fmaxf(s0, s1), fmaxf(s2, s3));
    float e0 = __expf(s0 - m), e1 = __expf(s1 - m), e2 = __expf(s2 - m), e3 = __expf(s3 - m);
    float inv = 1.f / (e0 + e1 + e2 + e3);
    sSc[b][h][0] = e0 * inv; sSc[b][h][1] = e1 * inv;
    sSc[b][h][2] = e2 * inv; sSc[b][h][3] = e3 * inv;
  }
  __syncthreads();

  _Float16* A2 = sAtt[0]; // [16][64]
#pragma unroll
  for (int i = 0; i < 8; ++i) {
    int idx = threadIdx.x + i * 128;
    int b = idx >> 6, col = idx & 63, h = col >> 5;
    float acc = 0.f;
#pragma unroll
    for (int jj = 0; jj < 4; ++jj)
      acc += sSc[b][h][jj] * (float)V2[(b * 4 + jj) * D64 + col];
    A2[idx] = (_Float16)acc;
  }
  __syncthreads();

  // O2 = A2 @ wmO^T + X
  float* O2 = sH1[1];
  {
    v16h a0 = load_frag_h(A2, D64), a1 = load_frag_h(A2 + 32, D64);
    int nt = wid;
    v8f acc = zero8();
    acc = wmma16(a0, load_frag_h(wH + OFF_WMO + nt * 16 * D64, D64), acc);
    acc = wmma16(a1, load_frag_h(wH + OFF_WMO + nt * 16 * D64 + 32, D64), acc);
#pragma unroll
    for (int r = 0; r < 8; ++r) {
      int row = rb + r;
      int gb = blockBase + row; if (gb >= NB) gb = NB - 1;
      O2[row * D64 + nt * 16 + col16] = acc[r] + bundles[(long long)gb * D64 + nt * 16 + col16];
    }
  }
  __syncthreads();

  float* H2 = sH1[2];
  _Float16* H2h = sHh[0];
  if (threadIdx.x < 16) {
    int row = threadIdx.x;
    const float4* hp = (const float4*)(O2 + row * D64);
    float mu = 0.f;
#pragma unroll
    for (int cb = 0; cb < 16; ++cb) { float4 v = hp[cb]; mu += v.x + v.y + v.z + v.w; }
    mu *= (1.f / D64);
    float var = 0.f;
#pragma unroll
    for (int cb = 0; cb < 16; ++cb) {
      float4 v = hp[cb];
      var += (v.x-mu)*(v.x-mu) + (v.y-mu)*(v.y-mu) + (v.z-mu)*(v.z-mu) + (v.w-mu)*(v.w-mu);
    }
    float rs = rsqrtf(var * (1.f / D64) + 1e-5f);
#pragma unroll
    for (int cb = 0; cb < 8; ++cb) {
      v8h pk;
#pragma unroll
      for (int i = 0; i < 8; ++i) {
        int c = cb * 8 + i;
        float v = (O2[row * D64 + c] - mu) * rs * wm_ln1g[c] + wm_ln1b[c];
        H2[row * D64 + c] = v;
        pk[i] = (_Float16)v;
      }
      *(v8h*)(H2h + row * D64 + cb * 8) = pk;
    }
  }
  __syncthreads();

  _Float16* T2 = sHh[1];
  {
    v16h a0 = load_frag_h(H2h, D64), a1 = load_frag_h(H2h + 32, D64);
    int nt = wid;
    v8f acc = zero8();
    acc = wmma16(a0, load_frag_h(wH + OFF_WMF1 + nt * 16 * D64, D64), acc);
    acc = wmma16(a1, load_frag_h(wH + OFF_WMF1 + nt * 16 * D64 + 32, D64), acc);
    float bb = wm_f1b[nt * 16 + col16];
#pragma unroll
    for (int r = 0; r < 8; ++r)
      T2[(rb + r) * D64 + nt * 16 + col16] = (_Float16)fmaxf(acc[r] + bb, 0.f);
  }
  __syncthreads();
  {
    v16h t0 = load_frag_h(T2, D64), t1 = load_frag_h(T2 + 32, D64);
    int nt = wid;
    v8f acc = zero8();
    acc = wmma16(t0, load_frag_h(wH + OFF_WMF2 + nt * 16 * D64, D64), acc);
    acc = wmma16(t1, load_frag_h(wH + OFF_WMF2 + nt * 16 * D64 + 32, D64), acc);
    float bb = wm_f2b[nt * 16 + col16];
#pragma unroll
    for (int r = 0; r < 8; ++r) {
      int row = rb + r;
      O2[row * D64 + nt * 16 + col16] = acc[r] + bb + H2[row * D64 + nt * 16 + col16];
    }
  }
  __syncthreads();

  if (threadIdx.x < 16) {
    int row = threadIdx.x;
    int gb = blockBase + row;
    const float4* hp = (const float4*)(O2 + row * D64);
    float mu = 0.f;
#pragma unroll
    for (int cb = 0; cb < 16; ++cb) { float4 v = hp[cb]; mu += v.x + v.y + v.z + v.w; }
    mu *= (1.f / D64);
    float var = 0.f;
#pragma unroll
    for (int cb = 0; cb < 16; ++cb) {
      float4 v = hp[cb];
      var += (v.x-mu)*(v.x-mu) + (v.y-mu)*(v.y-mu) + (v.z-mu)*(v.z-mu) + (v.w-mu)*(v.w-mu);
    }
    float rs = rsqrtf(var * (1.f / D64) + 1e-5f);
    if (gb < NB) {
#pragma unroll
      for (int cb = 0; cb < 16; ++cb) {
        float4 v = hp[cb];
        float4 o;
        o.x = (v.x - mu) * rs * wm_ln2g[cb*4+0] + wm_ln2b[cb*4+0];
        o.y = (v.y - mu) * rs * wm_ln2g[cb*4+1] + wm_ln2b[cb*4+1];
        o.z = (v.z - mu) * rs * wm_ln2g[cb*4+2] + wm_ln2b[cb*4+2];
        o.w = (v.w - mu) * rs * wm_ln2g[cb*4+3] + wm_ln2b[cb*4+3];
        *(float4*)(out + (long long)gb * D64 + cb * 4) = o;
      }
    }
  }
}

extern "C" void kernel_launch(void* const* d_in, const int* in_sizes, int n_in,
                              void* d_out, int out_size, void* d_ws, size_t ws_size,
                              hipStream_t stream) {
  const float* items   = (const float*)d_in[0];
  const float* bundles = (const float*)d_in[1];
  const float* bimp    = (const float*)d_in[2];
  const float* wi_q   = (const float*)d_in[3];
  const float* wi_k   = (const float*)d_in[4];
  const float* wi_v   = (const float*)d_in[5];
  const float* wi_o   = (const float*)d_in[6];
  const float* wi_ln1g = (const float*)d_in[7];
  const float* wi_ln1b = (const float*)d_in[8];
  const float* wi_ln2g = (const float*)d_in[9];
  const float* wi_ln2b = (const float*)d_in[10];
  const float* wi_f1w = (const float*)d_in[11];
  const float* wi_f1b = (const float*)d_in[12];
  const float* wi_f2w = (const float*)d_in[13];
  const float* wi_f2b = (const float*)d_in[14];
  const float* wm_q   = (const float*)d_in[15];
  const float* wm_k   = (const float*)d_in[16];
  const float* wm_v   = (const float*)d_in[17];
  const float* wm_o   = (const float*)d_in[18];
  const float* wm_ln1g = (const float*)d_in[19];
  const float* wm_ln1b = (const float*)d_in[20];
  const float* wm_ln2g = (const float*)d_in[21];
  const float* wm_ln2b = (const float*)d_in[22];
  const float* wm_f1w = (const float*)d_in[23];
  const float* wm_f1b = (const float*)d_in[24];
  const float* wm_f2w = (const float*)d_in[25];
  const float* wm_f2b = (const float*)d_in[26];
  const int* bitems = (const int*)d_in[27];

  const int NB = in_sizes[1] / D64;
  const int Kk = in_sizes[27] / NB;

  _Float16* wH = (_Float16*)d_ws;
  float* Qi = (float*)((char*)d_ws + (size_t)N_WHALF * sizeof(_Float16));

  int prepTot = N_WHALF + MIND * D64;
  isab_prep<<<(prepTot + 255) / 256, 256, 0, stream>>>(
      wi_k, wi_v, wi_o, wi_f1w, wi_f2w,
      wm_q, wm_k, wm_v, wm_o, wm_f1w, wm_f2w,
      wi_q, bimp, wH, Qi);

  int nblk = (NB + 15) / 16;
  isab_kernel<<<nblk, 128, 0, stream>>>(
      items, bundles, bimp, Qi, wH, bitems,
      wi_ln1g, wi_ln1b, wi_ln2g, wi_ln2b, wi_f1b, wi_f2b,
      wm_ln1g, wm_ln1b, wm_ln2g, wm_ln2b, wm_f1b, wm_f2b,
      (float*)d_out, NB, Kk);
}
